// SelfAttention_42984032698794
// MI455X (gfx1250) — matrix operand order repaired and checked by reading, not compiled, not run on hardware
//
#include <hip/hip_runtime.h>

// ---------------------------------------------------------------------------
// Types for CDNA5 WMMA (wave32): v16bf A/B operands, v8f C/D accumulators.
// ---------------------------------------------------------------------------
typedef __attribute__((ext_vector_type(8)))  float        v8f;
typedef __attribute__((ext_vector_type(16))) __bf16       v16bf;
typedef __attribute__((ext_vector_type(4)))  unsigned int u32x4;

union FragAB { v16bf bf; u32x4 u[2]; };
union FragC  { v8f v; float f[8]; };

__device__ __forceinline__ unsigned short to_bf16(float x) {
    unsigned int u = __float_as_uint(x);
    unsigned int r = u + 0x7FFFu + ((u >> 16) & 1u);   // round-to-nearest-even
    return (unsigned short)(r >> 16);
}
__device__ __forceinline__ float bf16_to_f32(unsigned short h) {
    return __uint_as_float(((unsigned int)h) << 16);
}

// LDS byte offset of a __shared__ element: low 32 bits of the flat address
// (generic LDS addresses are {aperture[63:32], offset[31:0]} on gfx1250).
__device__ __forceinline__ unsigned lds_off(const void* p) {
    return (unsigned)(unsigned long long)p;
}

// Async global -> LDS 16-byte copy (GVS mode: SGPR base + u32 VGPR byte offset).
// Tracked by ASYNCcnt; consumer must s_wait_asynccnt then barrier.
__device__ __forceinline__ void async_ld_b128(unsigned ldsaddr, unsigned voff,
                                              unsigned long long base) {
    asm volatile("global_load_async_to_lds_b128 %0, %1, %2"
                 :: "v"(ldsaddr), "v"(voff), "s"(base) : "memory");
}
__device__ __forceinline__ void wait_asynccnt0() {
    asm volatile("s_wait_asynccnt 0" ::: "memory");
}

// ---------------------------------------------------------------------------
// Elementwise f32 -> bf16 conversion
// ---------------------------------------------------------------------------
__global__ __launch_bounds__(256) void convert_bf16_kernel(
    const float* __restrict__ in, unsigned short* __restrict__ out, int n)
{
    int i = blockIdx.x * 256 + threadIdx.x;
    if (i < n) out[i] = to_bf16(in[i]);
}

// out[c][r] = bf16(in[r][c])   (in: R x C f32 row-major)
__global__ __launch_bounds__(256) void transpose_bf16_kernel(
    const float* __restrict__ in, unsigned short* __restrict__ out, int R, int C)
{
    int i = blockIdx.x * 256 + threadIdx.x;
    if (i >= R * C) return;
    int r = i / C, c = i % C;
    out[(size_t)c * R + r] = to_bf16(in[i]);
}

// ---------------------------------------------------------------------------
// Tiled bf16 WMMA GEMM:  out[M][N] = A[M][K] * Bt[N][K]^T + bias[N]
// Tile 128x128, K-step 32, 256 threads (8 waves), double-buffered LDS with
// async global->LDS staging. Each wave: 16 rows x 128 cols = 8 WMMA n-tiles.
// mode 0: f32 row-major out.  mode 1: bf16 -> [B,H,S,D].  mode 2: bf16 -> [B,H,D,S].
// ---------------------------------------------------------------------------
__global__ __launch_bounds__(256) void gemm_bf16_wmma_kernel(
    const unsigned short* __restrict__ A,
    const unsigned short* __restrict__ Bt,
    const float* __restrict__ bias,
    float* __restrict__ outF,
    unsigned short* __restrict__ outH,
    int M, int N, int K, int mode, int H, int D, int S)
{
    __shared__ alignas(16) unsigned short Al[2][128][32];
    __shared__ alignas(16) unsigned short Bl[2][128][32];

    const int t    = threadIdx.x;
    const int wv   = t >> 5;
    const int lane = t & 31;
    const int hl   = lane >> 4;   // which 16-lane half
    const int l16  = lane & 15;
    const int m0   = blockIdx.y * 128;
    const int n0   = blockIdx.x * 128;

    const unsigned long long aBase = (unsigned long long)A;
    const unsigned long long bBase = (unsigned long long)Bt;

    FragC acc[8];
#pragma unroll
    for (int nt = 0; nt < 8; ++nt)
#pragma unroll
        for (int i = 0; i < 8; ++i) acc[nt].f[i] = 0.0f;

    auto stage = [&](int buf, int k0) {
#pragma unroll
        for (int j = 0; j < 2; ++j) {
            int c   = t + 256 * j;       // 0..511 chunks of 8 bf16
            int row = c >> 2;
            int kc  = (c & 3) * 8;
            async_ld_b128(lds_off(&Al[buf][row][kc]),
                          (unsigned)(((unsigned)(m0 + row) * (unsigned)K +
                                      (unsigned)(k0 + kc)) * 2u), aBase);
            async_ld_b128(lds_off(&Bl[buf][row][kc]),
                          (unsigned)(((unsigned)(n0 + row) * (unsigned)K +
                                      (unsigned)(k0 + kc)) * 2u), bBase);
        }
    };

    stage(0, 0);
    wait_asynccnt0();
    __syncthreads();

    const int nK = K / 32;
    for (int kt = 0; kt < nK; ++kt) {
        int buf = kt & 1;
        if (kt + 1 < nK) stage(buf ^ 1, (kt + 1) * 32);

        // Preload A fragment + all 8 B fragments, then WMMAs back-to-back.
        FragAB afr;
        {
            int arow = wv * 16 + l16;
            int akc  = hl ? 8 : 0;
            afr.u[0] = *(const u32x4*)&Al[buf][arow][akc];
            afr.u[1] = *(const u32x4*)&Al[buf][arow][akc + 16];
        }
        FragAB bfr[8];
#pragma unroll
        for (int nt = 0; nt < 8; ++nt) {
            int brow = nt * 16 + l16;
            int bkc  = hl * 8;           // table order: K 8*hl+j, then 16+8*hl+j
            bfr[nt].u[0] = *(const u32x4*)&Bl[buf][brow][bkc];
            bfr[nt].u[1] = *(const u32x4*)&Bl[buf][brow][bkc + 16];
        }
#pragma unroll
        for (int nt = 0; nt < 8; ++nt) {
            acc[nt].v = __builtin_amdgcn_wmma_f32_16x16x32_bf16(
                false, afr.bf, false, bfr[nt].bf, (short)0, acc[nt].v, false, false);
        }
        wait_asynccnt0();
        __syncthreads();
    }

    // Epilogue. C layout: element i of v8f -> row = i + 8*hl, col = l16 (per n-tile)
#pragma unroll
    for (int nt = 0; nt < 8; ++nt) {
#pragma unroll
        for (int i = 0; i < 8; ++i) {
            int row = m0 + wv * 16 + hl * 8 + i;
            int col = n0 + nt * 16 + l16;
            float v = acc[nt].f[i] + bias[col];
            if (mode == 0) {
                outF[(size_t)row * N + col] = v;
            } else {
                int b = row / S, s = row % S;
                int h = col / D, d = col % D;
                size_t addr = (mode == 1)
                    ? ((size_t)(b * H + h) * S + s) * D + d
                    : ((size_t)(b * H + h) * D + d) * S + s;
                outH[addr] = to_bf16(v);
            }
        }
    }
}

// ---------------------------------------------------------------------------
// RoPE (in-place on bf16 [B,H,S,D]); optional 1/sqrt(D) scale folded into Q.
// ---------------------------------------------------------------------------
__global__ __launch_bounds__(256) void rope_kernel(
    unsigned short* __restrict__ q, int BHS, int S, int D, float scale)
{
    int half  = D >> 1;
    int idx   = blockIdx.x * 256 + threadIdx.x;
    int total = BHS * half;
    if (idx >= total) return;
    int bhs = idx / half;
    int i   = idx % half;
    int s   = bhs % S;

    float inv = __expf(-(float)(2 * i) * (9.210340372f / (float)D)); // 10000^(-2i/D)
    float ang = (float)s * inv;
    float sn, cs;
    __sincosf(ang, &sn, &cs);

    size_t base = (size_t)bhs * D;
    float x1 = bf16_to_f32(q[base + i]);
    float x2 = bf16_to_f32(q[base + i + half]);
    q[base + i]        = to_bf16((x1 * cs - x2 * sn) * scale);
    q[base + i + half] = to_bf16((x2 * cs + x1 * sn) * scale);
}

// ---------------------------------------------------------------------------
// Flash attention (causal). Grid: (S/128, B*H). 256 threads = 8 waves.
// Wave w owns query rows w*16..w*16+15 of the 128-query tile.
// Q: [B,H,S,D] bf16 (already scaled). K: [B,H,S,D]. V: [B,H,D,S] (transposed).
// ctx out: bf16 [B*S][H*D] row-major (A matrix of the final GEMM).
// ---------------------------------------------------------------------------
__global__ __launch_bounds__(256) void attn_kernel(
    const unsigned short* __restrict__ qg,
    const unsigned short* __restrict__ kg,
    const unsigned short* __restrict__ vg,
    unsigned short* __restrict__ ctx,
    int B, int H, int S, int D)
{
    __shared__ alignas(16) unsigned short QP[128][128];  // Q staging, then P
    __shared__ alignas(16) unsigned short Kl[64][128];
    __shared__ alignas(16) unsigned short Vtl[128][64];

    const int t    = threadIdx.x;
    const int wv   = t >> 5;
    const int lane = t & 31;
    const int hl   = lane >> 4;
    const int l16  = lane & 15;
    const int bh   = blockIdx.y;
    const int b    = bh / H, h = bh % H;
    const int q0   = blockIdx.x * 128;

    const unsigned headQK = (unsigned)bh * (unsigned)(S * D);
    const unsigned headV  = (unsigned)bh * (unsigned)(D * S);
    const unsigned long long qBase = (unsigned long long)qg;
    const unsigned long long kBase = (unsigned long long)kg;
    const unsigned long long vBase = (unsigned long long)vg;

    // Stage 128x128 Q tile (async -> LDS)
#pragma unroll
    for (int j = 0; j < 8; ++j) {
        int c = t + 256 * j, row = c >> 4, kc = (c & 15) * 8;
        async_ld_b128(lds_off(&QP[row][kc]),
                      (headQK + (unsigned)(q0 + row) * (unsigned)D + kc) * 2u, qBase);
    }
    wait_asynccnt0();
    __syncthreads();

    // Extract Q fragments (K-dim = D = 128 -> 4 k-steps), keep in registers
    FragAB qf[4];
    {
        int arow = wv * 16 + l16;
#pragma unroll
        for (int ks = 0; ks < 4; ++ks) {
            int kb = ks * 32 + (hl ? 8 : 0);
            qf[ks].u[0] = *(const u32x4*)&QP[arow][kb];
            qf[ks].u[1] = *(const u32x4*)&QP[arow][kb + 16];
        }
    }
    asm volatile("s_wait_dscnt 0" ::: "memory");  // reads done before QP reuse
    __syncthreads();

    FragC Oacc[8];
#pragma unroll
    for (int nt = 0; nt < 8; ++nt)
#pragma unroll
        for (int i = 0; i < 8; ++i) Oacc[nt].f[i] = 0.0f;
    float mrow[8], lrow[8];
#pragma unroll
    for (int i = 0; i < 8; ++i) { mrow[i] = -3.0e38f; lrow[i] = 0.0f; }

    const int nkv = (q0 + 128) / 64;   // causal: kv tiles with kv0 <= q0+127
    for (int kt = 0; kt < nkv; ++kt) {
        const int kv0 = kt * 64;
        // Stage K tile 64x128 and Vt tile 128x64 (async -> LDS)
#pragma unroll
        for (int j = 0; j < 4; ++j) {
            int c = t + 256 * j, row = c >> 4, kc = (c & 15) * 8;
            async_ld_b128(lds_off(&Kl[row][kc]),
                          (headQK + (unsigned)(kv0 + row) * (unsigned)D + kc) * 2u,
                          kBase);
        }
#pragma unroll
        for (int j = 0; j < 4; ++j) {
            int c = t + 256 * j, row = c >> 3, kc = (c & 7) * 8;
            async_ld_b128(lds_off(&Vtl[row][kc]),
                          (headV + (unsigned)row * (unsigned)S + kv0 + kc) * 2u,
                          vBase);
        }
        wait_asynccnt0();
        __syncthreads();

        // Scores S = Q * K^T : 16 rows x 64 kv cols (4 n-tiles)
        FragC Sa[4];
#pragma unroll
        for (int nt = 0; nt < 4; ++nt)
#pragma unroll
            for (int i = 0; i < 8; ++i) Sa[nt].f[i] = 0.0f;
#pragma unroll
        for (int ks = 0; ks < 4; ++ks) {
            FragAB bfr[4];
#pragma unroll
            for (int nt = 0; nt < 4; ++nt) {
                int brow = nt * 16 + l16;          // kv row (col of S)
                int kb   = ks * 32 + hl * 8;       // d index, table order
                bfr[nt].u[0] = *(const u32x4*)&Kl[brow][kb];
                bfr[nt].u[1] = *(const u32x4*)&Kl[brow][kb + 16];
            }
#pragma unroll
            for (int nt = 0; nt < 4; ++nt) {
                Sa[nt].v = __builtin_amdgcn_wmma_f32_16x16x32_bf16(
                    false, qf[ks].bf, false, bfr[nt].bf, (short)0, Sa[nt].v,
                    false, false);
            }
        }

        // Causal mask (only tiles overlapping the diagonal)
        if (kv0 + 64 > q0) {
#pragma unroll
            for (int nt = 0; nt < 4; ++nt) {
                int colg = kv0 + nt * 16 + l16;
#pragma unroll
                for (int i = 0; i < 8; ++i) {
                    int rowg = q0 + wv * 16 + hl * 8 + i;
                    if (colg > rowg) Sa[nt].f[i] = -3.0e38f;
                }
            }
        }

        // Online softmax: rows live in (vgpr index, 16-lane half); reduce over lanes
        float mnew[8], alpha[8];
#pragma unroll
        for (int i = 0; i < 8; ++i) {
            float mx = Sa[0].f[i];
#pragma unroll
            for (int nt = 1; nt < 4; ++nt) mx = fmaxf(mx, Sa[nt].f[i]);
            for (int off = 1; off < 16; off <<= 1)
                mx = fmaxf(mx, __shfl_xor(mx, off, 32));
            mnew[i]  = fmaxf(mrow[i], mx);
            alpha[i] = __expf(mrow[i] - mnew[i]);
            mrow[i]  = mnew[i];
        }
#pragma unroll
        for (int i = 0; i < 8; ++i) {
            float rs = 0.0f;
#pragma unroll
            for (int nt = 0; nt < 4; ++nt) {
                float p = __expf(Sa[nt].f[i] - mnew[i]);
                Sa[nt].f[i] = p;
                rs += p;
            }
            for (int off = 1; off < 16; off <<= 1)
                rs += __shfl_xor(rs, off, 32);
            lrow[i] = lrow[i] * alpha[i] + rs;
#pragma unroll
            for (int nt = 0; nt < 8; ++nt) Oacc[nt].f[i] *= alpha[i];
        }

        // Write P (bf16) to this wave's private LDS region, reload as A-operand
#pragma unroll
        for (int nt = 0; nt < 4; ++nt)
#pragma unroll
            for (int i = 0; i < 8; ++i)
                QP[wv * 16 + hl * 8 + i][nt * 16 + l16] = to_bf16(Sa[nt].f[i]);

        // O += P * V  (K-dim = 64 kv -> 2 k-steps; V^T gives contiguous kv)
#pragma unroll
        for (int ks = 0; ks < 2; ++ks) {
            FragAB pf;
            int prow = wv * 16 + l16;
            int kb   = ks * 32 + (hl ? 8 : 0);
            pf.u[0] = *(const u32x4*)&QP[prow][kb];
            pf.u[1] = *(const u32x4*)&QP[prow][kb + 16];
            FragAB vfr[8];
#pragma unroll
            for (int nt = 0; nt < 8; ++nt) {
                int vrow = nt * 16 + l16;            // d index (col of O)
                int vkb  = ks * 32 + hl * 8;         // kv, table order
                vfr[nt].u[0] = *(const u32x4*)&Vtl[vrow][vkb];
                vfr[nt].u[1] = *(const u32x4*)&Vtl[vrow][vkb + 16];
            }
#pragma unroll
            for (int nt = 0; nt < 8; ++nt) {
                Oacc[nt].v = __builtin_amdgcn_wmma_f32_16x16x32_bf16(
                    false, pf.bf, false, vfr[nt].bf, (short)0, Oacc[nt].v,
                    false, false);
            }
        }
        __syncthreads();
    }

    // Normalize and write ctx[b*S+row][h*D+d] (bf16)
#pragma unroll
    for (int i = 0; i < 8; ++i) {
        float inv = 1.0f / lrow[i];
        int rowg  = q0 + wv * 16 + hl * 8 + i;
        size_t rb = ((size_t)(b * S + rowg) * (H * D)) + (size_t)h * D;
#pragma unroll
        for (int nt = 0; nt < 8; ++nt)
            ctx[rb + nt * 16 + l16] = to_bf16(Oacc[nt].f[i] * inv);
    }
}

// ---------------------------------------------------------------------------
// Host-side orchestration
// ---------------------------------------------------------------------------
extern "C" void kernel_launch(void* const* d_in, const int* in_sizes, int n_in,
                              void* d_out, int out_size, void* d_ws, size_t ws_size,
                              hipStream_t stream)
{
    const int B = 2, S = 2048, F = 2048, H = 16, D = 128;
    const int M = B * S;       // 4096
    const int HD = H * D;      // 2048

    const float* x  = (const float*)d_in[0];
    // d_in[1] = causal mask (bool) -- implemented analytically
    const float* wq = (const float*)d_in[2];
    const float* bq = (const float*)d_in[3];
    const float* wk = (const float*)d_in[4];
    const float* bk = (const float*)d_in[5];
    const float* wv = (const float*)d_in[6];
    const float* bv = (const float*)d_in[7];
    const float* wo = (const float*)d_in[8];
    const float* bo = (const float*)d_in[9];
    float* out = (float*)d_out;

    char* ws = (char*)d_ws;
    size_t off = 0;
    auto wsalloc = [&](size_t bytes) -> void* {
        void* p = ws + off;
        off += (bytes + 255) & ~(size_t)255;
        return p;
    };
    unsigned short* xb   = (unsigned short*)wsalloc((size_t)M  * F  * 2);
    unsigned short* wqT  = (unsigned short*)wsalloc((size_t)HD * F  * 2);
    unsigned short* wkT  = (unsigned short*)wsalloc((size_t)HD * F  * 2);
    unsigned short* wvT  = (unsigned short*)wsalloc((size_t)HD * F  * 2);
    unsigned short* woT  = (unsigned short*)wsalloc((size_t)F  * HD * 2);
    unsigned short* qb   = (unsigned short*)wsalloc((size_t)M  * HD * 2);
    unsigned short* kb   = (unsigned short*)wsalloc((size_t)M  * HD * 2);
    unsigned short* vb   = (unsigned short*)wsalloc((size_t)M  * HD * 2);
    unsigned short* ctxb = (unsigned short*)wsalloc((size_t)M  * HD * 2);

    // 1) precision conversion / weight transposes
    convert_bf16_kernel<<<(M * F + 255) / 256, 256, 0, stream>>>(x, xb, M * F);
    transpose_bf16_kernel<<<(F * HD + 255) / 256, 256, 0, stream>>>(wq, wqT, F, HD);
    transpose_bf16_kernel<<<(F * HD + 255) / 256, 256, 0, stream>>>(wk, wkT, F, HD);
    transpose_bf16_kernel<<<(F * HD + 255) / 256, 256, 0, stream>>>(wv, wvT, F, HD);
    transpose_bf16_kernel<<<(HD * F + 255) / 256, 256, 0, stream>>>(wo, woT, HD, F);

    // 2) QKV projections (WMMA)
    dim3 gg(HD / 128, M / 128);   // (16, 32)
    gemm_bf16_wmma_kernel<<<gg, 256, 0, stream>>>(xb, wqT, bq, nullptr, qb,
                                                  M, HD, F, 1, H, D, S);
    gemm_bf16_wmma_kernel<<<gg, 256, 0, stream>>>(xb, wkT, bk, nullptr, kb,
                                                  M, HD, F, 1, H, D, S);
    gemm_bf16_wmma_kernel<<<gg, 256, 0, stream>>>(xb, wvT, bv, nullptr, vb,
                                                  M, HD, F, 2, H, D, S);

    // 3) RoPE (scale 1/sqrt(D) folded into Q)
    int BHS = B * H * S;
    int nrope = (BHS * (D / 2) + 255) / 256;
    rope_kernel<<<nrope, 256, 0, stream>>>(qb, BHS, S, D, 0.0883883476483184f);
    rope_kernel<<<nrope, 256, 0, stream>>>(kb, BHS, S, D, 1.0f);

    // 4) causal flash attention (WMMA)
    dim3 ga(S / 128, B * H);      // (16, 32)
    attn_kernel<<<ga, 256, 0, stream>>>(qb, kb, vb, ctxb, B, H, S, D);

    // 5) output projection (WMMA, f32 out + bias)
    gemm_bf16_wmma_kernel<<<gg, 256, 0, stream>>>(ctxb, woT, bo, out, nullptr,
                                                  M, F, HD, 0, H, D, S);
}
